// SEDD_81518479278610
// MI455X (gfx1250) — matrix operand order repaired and checked by reading, not compiled, not run on hardware
//
#include <hip/hip_runtime.h>
#include <cstdint>
#include <cstddef>

// ---------- model dims (fixed by setup_inputs) ----------
namespace cfg {
constexpr int B = 2, S = 1024, D = 768, H = 12, L = 12, V = 50258;
constexpr int F = 256, MLP = 3072, D3 = 3 * D, CA = 6 * D; // 2304, 4608
constexpr int DH = 64;                                     // head dim
}

// ---------- vector types ----------
typedef __attribute__((ext_vector_type(16))) __bf16    v16bf;
typedef __attribute__((ext_vector_type(8)))  float     v8f;
typedef __attribute__((ext_vector_type(4)))  uint32_t  u32x4;

union Frag {
  uint32_t u[8];
  u32x4    q[2];
  v16bf    v;
};

// ---------- helpers ----------
__device__ __forceinline__ uint32_t pack2(float a, float b) {
  union { __bf16 h[2]; uint32_t u; } z;
  z.h[0] = (__bf16)a; z.h[1] = (__bf16)b;
  return z.u;
}
__device__ __forceinline__ float silu_f(float x) { return x / (1.0f + __expf(-x)); }
__device__ __forceinline__ float gelu_f(float x) {
  return 0.5f * x * (1.0f + tanhf(0.7978845608028654f * (x + 0.044715f * x * x * x)));
}
__device__ __forceinline__ v8f wmma_bf16(const Frag& a, const Frag& b, v8f c) {
  return __builtin_amdgcn_wmma_f32_16x16x32_bf16(false, a.v, false, b.v,
                                                 (short)0, c, false, false);
}
// xor-butterfly shuffle within each 16-lane row via v_permlane16_b32 (VALU, no LDS)
__device__ __forceinline__ float plane16(float v, uint32_t slo, uint32_t shi) {
  uint32_t u = __builtin_bit_cast(uint32_t, v);
  uint32_t r = __builtin_amdgcn_permlane16(u, u, slo, shi, false, false);
  return __builtin_bit_cast(float, r);
}
__device__ __forceinline__ float rowmax16(float v) {
  v = fmaxf(v, plane16(v, 0x67452301u, 0xEFCDAB89u));   // xor 1
  v = fmaxf(v, plane16(v, 0x54761032u, 0xDCFE98BAu));   // xor 2
  v = fmaxf(v, plane16(v, 0x32107654u, 0xBA98FEDCu));   // xor 4
  v = fmaxf(v, plane16(v, 0xFEDCBA98u, 0x76543210u));   // xor 8
  return v;
}

// ============================================================
// 0) fp32 -> bf16 weight conversion (once per launch per matrix)
// ============================================================
__global__ void k_w2bf(const float* __restrict__ src, uint32_t* __restrict__ dst, int n4) {
  int i = blockIdx.x * blockDim.x + threadIdx.x;
  if (i >= n4) return;
  float4 f = *(const float4*)(src + 4 * (size_t)i);
  dst[2 * i]     = pack2(f.x, f.y);
  dst[2 * i + 1] = pack2(f.z, f.w);
}

// ============================================================
// 1) timestep embedding + t-MLP:  c = silu(silu(tf@W1^T+b1)@W2^T+b2)
// ============================================================
__global__ void k_temb(const float* __restrict__ sigma,
                       const float* __restrict__ w1, const float* __restrict__ b1,
                       const float* __restrict__ w2, const float* __restrict__ b2,
                       float* __restrict__ c) {
  using namespace cfg;
  __shared__ float tf[F];
  __shared__ float hdn[D];
  const int b = blockIdx.x, t = threadIdx.x;
  if (t < F / 2) {
    float a = sigma[b] * __expf(-9.210340371976184f * (float)t / (float)(F / 2));
    tf[t] = __cosf(a);
    tf[t + F / 2] = __sinf(a);
  }
  __syncthreads();
  for (int j = t; j < D; j += blockDim.x) {
    const float* wr = w1 + (size_t)j * F;
    float s = 0.f;
    for (int k = 0; k < F; ++k) s += tf[k] * wr[k];
    hdn[j] = silu_f(s + b1[j]);
  }
  __syncthreads();
  for (int i = t; i < D; i += blockDim.x) {
    const float* wr = w2 + (size_t)i * D;
    float s = 0.f;
    for (int k = 0; k < D; ++k) s += hdn[k] * wr[k];
    c[b * D + i] = silu_f(s + b2[i]);
  }
}

// ============================================================
// 2) embedding gather
// ============================================================
__global__ void k_embed(const int* __restrict__ idx, const float* __restrict__ emb,
                        float* __restrict__ x, int n) {
  int i = blockIdx.x * blockDim.x + threadIdx.x;
  if (i >= n) return;
  int r = i / cfg::D, col = i - r * cfg::D;
  x[i] = emb[(size_t)idx[r] * cfg::D + col];
}

// ============================================================
// 3) adaLN projection: ada[b][j] = c[b]·W[j] + bias[j]
// ============================================================
__global__ void k_ada(const float* __restrict__ c, const float* __restrict__ W,
                      const float* __restrict__ bias, float* __restrict__ ada,
                      int outD, int nB) {
  int i = blockIdx.x * blockDim.x + threadIdx.x;
  if (i >= nB * outD) return;
  int b = i / outD, j = i - b * outD;
  const float* cr = c + b * cfg::D;
  const float* wr = W + (size_t)j * cfg::D;
  float s = 0.f;
  for (int k = 0; k < cfg::D; ++k) s += cr[k] * wr[k];
  ada[i] = s + bias[j];
}

// ============================================================
// 4) LayerNorm + modulate -> bf16
// ============================================================
__global__ void k_ln_mod(const float* __restrict__ x, const float* __restrict__ w,
                         const float* __restrict__ ada, int sh_off, int sc_off,
                         int ada_stride, __bf16* __restrict__ out) {
  using namespace cfg;
  __shared__ float sred[256];
  const int row = blockIdx.x, tid = threadIdx.x, b = row / S;
  const float* xr = x + (size_t)row * D;
  float s = 0.f;
  for (int i = tid; i < D; i += 256) s += xr[i];
  sred[tid] = s; __syncthreads();
  for (int o = 128; o > 0; o >>= 1) { if (tid < o) sred[tid] += sred[tid + o]; __syncthreads(); }
  const float mu = sred[0] * (1.0f / D);
  __syncthreads();
  float v = 0.f;
  for (int i = tid; i < D; i += 256) { float d = xr[i] - mu; v += d * d; }
  sred[tid] = v; __syncthreads();
  for (int o = 128; o > 0; o >>= 1) { if (tid < o) sred[tid] += sred[tid + o]; __syncthreads(); }
  const float rs = rsqrtf(sred[0] * (1.0f / D) + 1e-5f);
  const float* ab = ada + (size_t)b * ada_stride;
  for (int i = tid; i < D; i += 256) {
    float y = (xr[i] - mu) * rs * w[i];
    y = y * (1.0f + ab[sc_off + i]) + ab[sh_off + i];
    out[(size_t)row * D + i] = (__bf16)y;
  }
}

// ============================================================
// 5) WMMA GEMM (all-bf16 operands): out[M×N] = A[M×K] @ W[N×K]^T (+bias)
//    one wave per 32 x (NT*16) tile: 2*NT wmma per K-step
//    EPI: 0=f32, 1=bf16, 2=gelu->bf16, 3=x += gate*acc
//    TAIL: column-overrun masking (branch-free, no EXEC writes)
// ============================================================
template <int EPI, bool TAIL, int NT>
__global__ void k_gemm(const __bf16* __restrict__ A, const __bf16* __restrict__ W,
                       const float* __restrict__ bias, void* __restrict__ out,
                       float* __restrict__ xres, const float* __restrict__ gate,
                       int M, int N, int K, int gstride, int srows,
                       int totalTiles, int nmac) {
  const int wid = blockIdx.x * (blockDim.x >> 5) + (threadIdx.x >> 5);
  if (wid >= totalTiles) return;                       // wave-uniform
  const int lane = threadIdx.x & 31, lm = lane & 15, h = lane >> 4;
  const int mt = wid / nmac, nm = wid - mt * nmac;
  const int row0 = mt * 32, col0 = nm * (NT * 16);

  v8f acc[2][NT] = {};
  for (int k0 = 0; k0 < K; k0 += 32) {
    Frag a[2];
#pragma unroll
    for (int mi = 0; mi < 2; ++mi) {
      const __bf16* ar = A + (size_t)(row0 + mi * 16 + lm) * K + k0 + (h << 3);
      a[mi].q[0] = *(const u32x4*)ar;          // elems [h*8, h*8+8)
      a[mi].q[1] = *(const u32x4*)(ar + 16);   // elems [16+h*8, ...)
    }
#pragma unroll
    for (int nt = 0; nt < NT; ++nt) {
      const int col = col0 + nt * 16 + lm;
      const int cc = TAIL ? (col < N ? col : N - 1) : col;
      const __bf16* wr = W + (size_t)cc * K + k0 + (h << 3);
      Frag b;
      b.q[0] = *(const u32x4*)wr;              // elems [h*8, h*8+8)
      b.q[1] = *(const u32x4*)(wr + 16);       // elems [16+h*8, ...)
      __builtin_prefetch((const void*)(wr + 32), 0, 3);  // speculative, next K-step
      if (TAIL) {
        const uint32_t mk = (col < N) ? 0xFFFFFFFFu : 0u;   // v_cndmask only
#pragma unroll
        for (int i = 0; i < 8; ++i) b.u[i] &= mk;
      }
#pragma unroll
      for (int mi = 0; mi < 2; ++mi)
        acc[mi][nt] = wmma_bf16(a[mi], b, acc[mi][nt]);
    }
  }
  // epilogue (after all WMMA; per-lane divergence OK here)
#pragma unroll
  for (int mi = 0; mi < 2; ++mi)
#pragma unroll
    for (int nt = 0; nt < NT; ++nt) {
      const int col = col0 + nt * 16 + lm;
      if (TAIL && col >= N) continue;
      const float bv = bias ? bias[col] : 0.f;
#pragma unroll
      for (int r = 0; r < 8; ++r) {
        const int row = row0 + mi * 16 + r + (h << 3);
        float v = acc[mi][nt][r] + bv;
        if (EPI == 0) {
          ((float*)out)[(size_t)row * N + col] = v;
        } else if (EPI == 1) {
          ((__bf16*)out)[(size_t)row * N + col] = (__bf16)v;
        } else if (EPI == 2) {
          ((__bf16*)out)[(size_t)row * N + col] = (__bf16)gelu_f(v);
        } else {
          const float g = gate[(size_t)(row / srows) * gstride + col];
          xres[(size_t)row * N + col] += g * v;
        }
      }
    }
}

// ============================================================
// 6) V transpose: vT[(b*H+h)*64 + d][s] = qkv[b,s, 2D + h*64 + d]
// ============================================================
__global__ void k_vtrans(const __bf16* __restrict__ qkv, __bf16* __restrict__ vT, int n) {
  using namespace cfg;
  int i = blockIdx.x * blockDim.x + threadIdx.x;
  if (i >= n) return;
  const int s = i & (S - 1);
  const int rest = i >> 10;          // S = 1024
  const int d = rest & (DH - 1);
  const int bh = rest >> 6;          // b*H + h
  const int b = bh / H, hh = bh - b * H;
  vT[i] = qkv[(size_t)(b * S + s) * D3 + 2 * D + hh * DH + d];
}

// ============================================================
// 7) Flash attention: one wave per (b, h, 16-row q-tile)
//    32-token k-tiles: QK^T (4 wmma) + online softmax + PV (4 wmma)
//    row-max: v_permlane16 butterfly; row-sum: P @ ones via WMMA
// ============================================================
__global__ void k_attn(const __bf16* __restrict__ qkv, const __bf16* __restrict__ vT,
                       __bf16* __restrict__ obuf) {
  using namespace cfg;
  constexpr int QT = S / 16;
  const int bid = blockIdx.x;
  const int qt = bid % QT;
  const int hh = (bid / QT) % H;
  const int b  = bid / (QT * H);
  const int lane = threadIdx.x & 31, lm = lane & 15, hf = lane >> 4;

  __shared__ float pl[16 * 32];

  // Q tile as two A fragments (dh = 64 -> 2 K-steps), 2 x b128 each
  Frag aq[2];
  const __bf16* qr = qkv + (size_t)(b * S + qt * 16 + lm) * D3 + hh * DH;
#pragma unroll
  for (int ks = 0; ks < 2; ++ks) {
    const __bf16* p = qr + ks * 32 + (hf << 3);
    aq[ks].q[0] = *(const u32x4*)p;
    aq[ks].q[1] = *(const u32x4*)(p + 16);
  }

  // B fragment of all 1.0 (bf16) for row-sum WMMA
  Frag ones;
#pragma unroll
  for (int j = 0; j < 8; ++j) ones.u[j] = 0x3F803F80u;

  float m_[8], l_[8], alpha[8];
#pragma unroll
  for (int r = 0; r < 8; ++r) { m_[r] = -1e30f; l_[r] = 0.f; }
  v8f oacc[4] = {};

  const __bf16* vbase = vT + (size_t)(b * H + hh) * DH * S;

  for (int kt = 0; kt < S / 32; ++kt) {
    const int tok0 = kt * 32;
    // ---- scores = Q K^T (two 16-col tiles) ----
    v8f sc[2] = {};
#pragma unroll
    for (int ct = 0; ct < 2; ++ct) {
#pragma unroll
      for (int ks = 0; ks < 2; ++ks) {
        Frag bk;
        const __bf16* kr = qkv + (size_t)(b * S + tok0 + ct * 16 + lm) * D3
                           + D + hh * DH + ks * 32 + (hf << 3);
        bk.q[0] = *(const u32x4*)kr;
        bk.q[1] = *(const u32x4*)(kr + 16);
        sc[ct] = wmma_bf16(aq[ks], bk, sc[ct]);
      }
    }
    // ---- online softmax: max via permlane16 butterfly (no LDS) ----
    float p0[8], p1[8];
#pragma unroll
    for (int r = 0; r < 8; ++r) {
      const float v0 = sc[0][r] * 0.125f, v1 = sc[1][r] * 0.125f;
      const float mx = rowmax16(fmaxf(v0, v1));
      const float mn = fmaxf(m_[r], mx);
      alpha[r] = __expf(m_[r] - mn);
      p0[r] = __expf(v0 - mn);
      p1[r] = __expf(v1 - mn);
      m_[r] = mn;
#pragma unroll
      for (int dt = 0; dt < 4; ++dt) oacc[dt][r] *= alpha[r];
    }
    // ---- P (D-layout f32) -> LDS -> A-fragment bf16 ----
#pragma unroll
    for (int r = 0; r < 8; ++r) {
      pl[(r + (hf << 3)) * 32 + lm] = p0[r];
      pl[(r + (hf << 3)) * 32 + 16 + lm] = p1[r];
    }
    __syncthreads();
    Frag ap;
#pragma unroll
    for (int j = 0; j < 8; ++j) {
      int ko = ((j & 4) << 2) + (hf << 3) + ((j & 3) << 1);
      ap.u[j] = pack2(pl[lm * 32 + ko], pl[lm * 32 + ko + 1]);
    }
    __syncthreads();
    // ---- row-sum via matrix pipe: rsum = P @ ones ----
    v8f rsum = wmma_bf16(ap, ones, v8f{});
#pragma unroll
    for (int r = 0; r < 8; ++r) l_[r] = l_[r] * alpha[r] + rsum[r];
    // ---- O += P @ V  (V pre-transposed: 2 x b128 per fragment) ----
#pragma unroll
    for (int dt = 0; dt < 4; ++dt) {
      Frag bv;
      const __bf16* vr = vbase + (size_t)(dt * 16 + lm) * S + tok0 + (hf << 3);
      bv.q[0] = *(const u32x4*)vr;
      bv.q[1] = *(const u32x4*)(vr + 16);
      oacc[dt] = wmma_bf16(ap, bv, oacc[dt]);
    }
  }
  // ---- normalize + store bf16 ----
#pragma unroll
  for (int dt = 0; dt < 4; ++dt)
#pragma unroll
    for (int r = 0; r < 8; ++r) {
      const float v = oacc[dt][r] / l_[r];
      const size_t row = (size_t)(b * S + qt * 16 + r + (hf << 3));
      obuf[row * D + hh * DH + dt * 16 + lm] = (__bf16)v;
    }
}

// ============================================================
// host-side orchestration
// ============================================================
static inline void launch_w2bf(hipStream_t st, const float* src, __bf16* dst, size_t n) {
  const int n4 = (int)(n / 4);
  k_w2bf<<<(n4 + 255) / 256, 256, 0, st>>>(src, (uint32_t*)dst, n4);
}

template <int EPI, bool TAIL, int NT>
static inline void launch_gemm(hipStream_t st, const __bf16* A, const __bf16* W,
                               const float* bias, void* out, int M, int N, int K,
                               float* xres, const float* gate, int gstride, int srows) {
  const int macw = NT * 16;
  const int nmac = (N + macw - 1) / macw;
  const int tiles = (M / 32) * nmac;
  const int blocks = (tiles + 7) / 8;     // 8 waves (256 threads) per block
  k_gemm<EPI, TAIL, NT><<<blocks, 256, 0, st>>>(A, W, bias, out, xres, gate,
                                                M, N, K, gstride, srows, tiles, nmac);
}

extern "C" void kernel_launch(void* const* d_in, const int* /*in_sizes*/, int /*n_in*/,
                              void* d_out, int /*out_size*/, void* d_ws, size_t /*ws_size*/,
                              hipStream_t stream) {
  using namespace cfg;
  const int*   indices = (const int*)  d_in[0];
  const float* sigma   = (const float*)d_in[1];
  const float* embed   = (const float*)d_in[2];
  const float* t_w1    = (const float*)d_in[3];
  const float* t_b1    = (const float*)d_in[4];
  const float* t_w2    = (const float*)d_in[5];
  const float* t_b2    = (const float*)d_in[6];
  const float* norm1_w = (const float*)d_in[7];
  const float* qkv_w   = (const float*)d_in[8];
  const float* out_w   = (const float*)d_in[9];
  const float* norm2_w = (const float*)d_in[10];
  const float* mlp_w1  = (const float*)d_in[11];
  const float* mlp_b1  = (const float*)d_in[12];
  const float* mlp_w2  = (const float*)d_in[13];
  const float* mlp_b2  = (const float*)d_in[14];
  const float* ada_w   = (const float*)d_in[15];
  const float* ada_b   = (const float*)d_in[16];
  const float* normf_w = (const float*)d_in[17];
  const float* lin_w   = (const float*)d_in[18];
  const float* lin_b   = (const float*)d_in[19];
  const float* adaf_w  = (const float*)d_in[20];
  const float* adaf_b  = (const float*)d_in[21];

  // workspace carve-out (all offsets stay 256B aligned)
  char* p = (char*)d_ws;
  float*  c    = (float*)p;   p += (size_t)B * D * 4;
  float*  ada  = (float*)p;   p += (size_t)B * CA * 4;
  float*  x    = (float*)p;   p += (size_t)B * S * D * 4;
  __bf16* xm   = (__bf16*)p;  p += (size_t)B * S * D * 2;
  __bf16* qkvb = (__bf16*)p;  p += (size_t)B * S * D3 * 2;
  __bf16* obuf = (__bf16*)p;  p += (size_t)B * S * D * 2;
  __bf16* hbuf = (__bf16*)p;  p += (size_t)B * S * MLP * 2;
  __bf16* vT   = (__bf16*)p;  p += (size_t)B * S * D * 2;
  __bf16* wbf  = (__bf16*)p;  p += (size_t)V * D * 2;   // staging, max = lin_w

  const int M = B * S;

  k_temb<<<B, 256, 0, stream>>>(sigma, t_w1, t_b1, t_w2, t_b2, c);
  k_embed<<<(M * D + 255) / 256, 256, 0, stream>>>(indices, embed, x, M * D);

  for (int l = 0; l < L; ++l) {
    k_ada<<<(B * CA + 255) / 256, 256, 0, stream>>>(c, ada_w + (size_t)l * CA * D,
                                                    ada_b + (size_t)l * CA, ada, CA, B);
    // ---- MSA branch ----
    k_ln_mod<<<M, 256, 0, stream>>>(x, norm1_w + l * D, ada, 0 * D, 1 * D, CA, xm);
    launch_w2bf(stream, qkv_w + (size_t)l * D3 * D, wbf, (size_t)D3 * D);
    launch_gemm<1, false, 8>(stream, xm, wbf, nullptr, qkvb, M, D3, D,
                             nullptr, nullptr, 0, S);
    k_vtrans<<<(M * D + 255) / 256, 256, 0, stream>>>(qkvb, vT, M * D);
    k_attn<<<B * H * (S / 16), 32, 0, stream>>>(qkvb, vT, obuf);
    launch_w2bf(stream, out_w + (size_t)l * D * D, wbf, (size_t)D * D);
    launch_gemm<3, false, 4>(stream, obuf, wbf, nullptr, nullptr, M, D, D,
                             x, ada + 2 * D, CA, S);
    // ---- MLP branch ----
    k_ln_mod<<<M, 256, 0, stream>>>(x, norm2_w + l * D, ada, 3 * D, 4 * D, CA, xm);
    launch_w2bf(stream, mlp_w1 + (size_t)l * MLP * D, wbf, (size_t)MLP * D);
    launch_gemm<2, false, 8>(stream, xm, wbf, mlp_b1 + (size_t)l * MLP, hbuf,
                             M, MLP, D, nullptr, nullptr, 0, S);
    launch_w2bf(stream, mlp_w2 + (size_t)l * D * MLP, wbf, (size_t)D * MLP);
    launch_gemm<3, false, 4>(stream, hbuf, wbf, mlp_b2 + (size_t)l * D, nullptr,
                             M, D, MLP, x, ada + 5 * D, CA, S);
  }

  // ---- final adaLN + logits ----
  k_ada<<<(B * 2 * D + 255) / 256, 256, 0, stream>>>(c, adaf_w, adaf_b, ada, 2 * D, B);
  k_ln_mod<<<M, 256, 0, stream>>>(x, normf_w, ada, 0 * D, 1 * D, 2 * D, xm);
  launch_w2bf(stream, lin_w, wbf, (size_t)V * D);
  launch_gemm<0, true, 8>(stream, xm, wbf, lin_b, d_out, M, V, D,
                          nullptr, nullptr, 0, S);
}
